// PeriodicEmbedding_17892833755540
// MI455X (gfx1250) — compile-verified, matrix operand order checked by reading, not run on hardware
//
#include <hip/hip_runtime.h>
#include <hip/hip_bf16.h>

typedef __attribute__((ext_vector_type(16))) _Float16 v16h;
typedef __attribute__((ext_vector_type(8)))  float    v8f;
typedef __attribute__((ext_vector_type(4)))  int      v4i;

#define NTOK 256
#define DIM  64
#define HDIM 32
#define WAVES_PER_BLOCK 8
#define ROWS_PER_WAVE   64          // 4 M-tiles of 16
#define ROWS_PER_BLOCK  (WAVES_PER_BLOCK * ROWS_PER_WAVE)  // 512
#define LDSB_STRIDE 72              // 64 + 8 halves pad -> conflict-free b128 reads

// --- raw hardware trig: V_SIN/COS_F32 take input in revolutions (x/2pi) ---
#if __has_builtin(__builtin_amdgcn_sinf) && __has_builtin(__builtin_amdgcn_cosf)
  #define FAST_SIN(x) __builtin_amdgcn_sinf(x)
  #define FAST_COS(x) __builtin_amdgcn_cosf(x)
  #define TRIG_SCALE  1.0f                  // hw computes sin(2*pi*arg)
#else
  #define FAST_SIN(x) __sinf(x)
  #define FAST_COS(x) __cosf(x)
  #define TRIG_SCALE  6.283185307179586f    // pass radians
#endif

// --- CDNA5 async global->LDS staging (ASYNCcnt path), guarded ---
#if defined(__gfx1250__) && __has_builtin(__builtin_amdgcn_global_load_async_to_lds_b128)
  #define HAVE_ASYNC_LDS 1
#else
  #define HAVE_ASYNC_LDS 0
#endif

#if HAVE_ASYNC_LDS
// Builtin prototype (from hipcc diagnostic): param0 = int4 in AS1 (__device__),
// param1 = int4 in AS3 (__shared__), then imm offset, imm cpol.
typedef __attribute__((address_space(1))) v4i gv4i;
typedef __attribute__((address_space(3))) v4i lv4i;
#endif

__global__ __launch_bounds__(256, 2)
void periodic_embedding_wmma(const float* __restrict__ inputs,   // [B, NTOK]
                             const float* __restrict__ coef,     // [NTOK, HDIM]
                             const float* __restrict__ weight,   // [NTOK, DIM, DIM]
                             const float* __restrict__ bias,     // [NTOK, DIM]
                             float* __restrict__ out,            // [B, NTOK*DIM]
                             int B)
{
    const int n        = blockIdx.x;                  // token
    const int rowBlock = blockIdx.y * ROWS_PER_BLOCK; // base row in B
    const int tid      = threadIdx.x;
    const int wave     = tid >> 5;
    const int lane     = tid & 31;
    const int lhalf    = lane >> 4;   // 0: lanes 0-15, 1: lanes 16-31
    const int l16      = lane & 15;

    __shared__ _Float16 sW[DIM][LDSB_STRIDE];  // transposed weights: sW[dout][k], f16
    __shared__ float    sBias[DIM];
    __shared__ float    sCoef[HDIM];

    // ---- stage coef / bias ----
    if (tid < HDIM) sCoef[tid] = coef[n * HDIM + tid];
    if (tid < DIM)  sBias[tid] = bias[n * DIM + tid];

    const float* Wg = weight + (size_t)n * (DIM * DIM);

#if HAVE_ASYNC_LDS
    // ---- async DMA of raw f32 W[n] (16KB) into LDS, then repack ----
    __shared__ float sWstage[DIM * DIM];
    {
        char* gbase = (char*)Wg;             // const dropped: builtin takes non-const
        char* lbase = (char*)&sWstage[0];
        #pragma unroll
        for (int i = 0; i < 4; ++i) {
            int boff = tid * 16 + i * 4096;
            __builtin_amdgcn_global_load_async_to_lds_b128(
                (gv4i*)(gbase + boff), (lv4i*)(lbase + boff), 0, 0);
        }
#if __has_builtin(__builtin_amdgcn_s_wait_asynccnt)
        __builtin_amdgcn_s_wait_asynccnt(0);
#else
        asm volatile("s_wait_asynccnt 0" ::: "memory");
#endif
    }
    __syncthreads();
    // repack: f32 [k][dout] -> f16 transposed sW[dout][k]
    #pragma unroll
    for (int i = 0; i < 4; ++i) {
        int idx = tid * 4 + i * 1024;
        float4 w4 = *(const float4*)(&sWstage[idx]);
        int k  = idx >> 6;
        int dc = idx & 63;
        sW[dc + 0][k] = (_Float16)w4.x;
        sW[dc + 1][k] = (_Float16)w4.y;
        sW[dc + 2][k] = (_Float16)w4.z;
        sW[dc + 3][k] = (_Float16)w4.w;
    }
#else
    // ---- direct global stage: f32 [k][dout] -> f16 transposed sW[dout][k] ----
    #pragma unroll
    for (int i = 0; i < 4; ++i) {
        int idx = tid * 4 + i * 1024;
        float4 w4 = *(const float4*)(Wg + idx);
        int k  = idx >> 6;
        int dc = idx & 63;
        sW[dc + 0][k] = (_Float16)w4.x;
        sW[dc + 1][k] = (_Float16)w4.y;
        sW[dc + 2][k] = (_Float16)w4.z;
        sW[dc + 3][k] = (_Float16)w4.w;
    }
#endif
    __syncthreads();

    // ---- per-lane coef values in the A-fragment K order ----
    // A 16-bit layout (16x32): lanes 0-15 hold K={0..7,16..23}, lanes 16-31 K={8..15,24..31}
    float cf[16];
    #pragma unroll
    for (int j = 0; j < 16; ++j) {
        int h = (j < 8 ? j : j + 8) + lhalf * 8;   // K index within 32-chunk == coef index
        cf[j] = TRIG_SCALE * sCoef[h];
    }

    // ---- B-operand fragments: 4 N-tiles x 2 K-chunks, built once, reused 4 M-tiles ----
    // B layout mirrors A: lane column = l16, lane half selects K offset 0/8 within chunk.
    v16h Bt[4][2];
    #pragma unroll
    for (int ct = 0; ct < 4; ++ct) {
        const _Float16* wr = &sW[ct * 16 + l16][0];
        #pragma unroll
        for (int kc = 0; kc < 2; ++kc) {
            const int off = lhalf * 8 + kc * 32;
            #pragma unroll
            for (int j = 0; j < 8; ++j) {
                Bt[ct][kc][j]     = wr[off + j];        // K = off+j      (runs 0..7)
                Bt[ct][kc][j + 8] = wr[off + 16 + j];   // K = off+16+j   (runs 16..23)
            }
        }
    }

    const size_t outRowStride = (size_t)NTOK * DIM;

    // ---- 4 M-tiles of 16 rows per wave ----
    for (int m = 0; m < 4; ++m) {
        const int rbase = rowBlock + wave * ROWS_PER_WAVE + m * 16;

        // A row for this lane: M = l16 (both lane halves); read-once -> non-temporal
        const float x =
            __builtin_nontemporal_load(&inputs[(size_t)(rbase + l16) * NTOK + n]);

        v16h A0, A1;   // K-chunk 0..31 -> sin ; K-chunk 32..63 -> cos
        #pragma unroll
        for (int j = 0; j < 16; ++j) {
            float v = cf[j] * x;
            A0[j] = (_Float16)FAST_SIN(v);
            A1[j] = (_Float16)FAST_COS(v);
        }

        v8f acc[4] = {v8f{}, v8f{}, v8f{}, v8f{}};
        #pragma unroll
        for (int ct = 0; ct < 4; ++ct) {
            acc[ct] = __builtin_amdgcn_wmma_f32_16x16x32_f16(
                false, A0, false, Bt[ct][0], (short)0, acc[ct], false, false);
            acc[ct] = __builtin_amdgcn_wmma_f32_16x16x32_f16(
                false, A1, false, Bt[ct][1], (short)0, acc[ct], false, false);
        }

        // ---- bias + relu + streamed (non-temporal) store ----
        // C/D layout: VGPR i -> M = i (lanes 0-15), M = i+8 (lanes 16-31); N = l16
        #pragma unroll
        for (int ct = 0; ct < 4; ++ct) {
            const float bv = sBias[ct * 16 + l16];
            #pragma unroll
            for (int i = 0; i < 8; ++i) {
                int row = rbase + i + lhalf * 8;
                float val = acc[ct][i] + bv;
                val = fmaxf(val, 0.0f);
                __builtin_nontemporal_store(
                    val, &out[(size_t)row * outRowStride + (size_t)n * DIM + ct * 16 + l16]);
            }
        }
    }
}

extern "C" void kernel_launch(void* const* d_in, const int* in_sizes, int n_in,
                              void* d_out, int out_size, void* d_ws, size_t ws_size,
                              hipStream_t stream) {
    const float* inputs = (const float*)d_in[0];   // [B, 256]
    const float* coef   = (const float*)d_in[1];   // [1, 256, 32]
    const float* weight = (const float*)d_in[2];   // [1, 256, 64, 64]
    const float* bias   = (const float*)d_in[3];   // [1, 256, 64]
    float*       out    = (float*)d_out;           // [B, 256*64]

    const int B = in_sizes[0] / NTOK;              // 8192

    dim3 grid(NTOK, B / ROWS_PER_BLOCK, 1);        // 256 x 16
    dim3 block(256, 1, 1);
    periodic_embedding_wmma<<<grid, block, 0, stream>>>(inputs, coef, weight, bias, out, B);
}
